// TrajPredictor_41738492182864
// MI455X (gfx1250) — matrix operand order repaired and checked by reading, not compiled, not run on hardware
//
#include <hip/hip_runtime.h>

// ---------------------------------------------------------------------------
// TrajPredictor for MI455X (gfx1250, wave32, WMMA).
// Phases:
//   1) pack_b_kernel     : f32 weights -> fragment-ready f16 (B-matrix layout)
//   2) agent_attn_kernel : per-(b,t) agent attention -> timestep reprs (B,T,3)
//   3) encoder_kernel    : GRU over T; 16-batch-row tile per WG, h in LDS,
//                          gates via v_wmma_f32_16x16x32_f16; enc_out -> f16
//   4) encproj_kernel    : (B*T,256)@(256,256) WMMA GEMM -> enc_proj f16
//   5) decoder_kernel    : 16 steps: dproj WMMA, streaming online-softmax
//                          temporal attention, GRU gates WMMA, pred head
// ---------------------------------------------------------------------------

#define HD 256
#define TT 256
#define G3 768
#define BM 16
#define PRED_LEN 16
#define NB 1024

typedef _Float16 h16_t;
typedef __attribute__((ext_vector_type(16))) _Float16 v16h;
typedef __attribute__((ext_vector_type(8)))  _Float16 v8h;
typedef __attribute__((ext_vector_type(8)))  float    v8f;

union V16H { v16h v; v8h h[2]; };

// A-fragment (16x32 f16, row-major source with leading dim ld):
// lanes 0-15: rows 0-15, halves 0..7 -> K=kb*32+0..7,  halves 8..15 -> K=+16..23
// lanes16-31: rows 0-15, halves 0..7 -> K=kb*32+8..15, halves 8..15 -> K=+24..31
__device__ __forceinline__ v16h load_a(const h16_t* base, int ld, int row,
                                       int kb, int grp) {
  const h16_t* p = base + row * ld + kb * 32 + grp * 8;
  V16H u;
  u.h[0] = *(const v8h*)(p);
  u.h[1] = *(const v8h*)(p + 16);
  return u.v;
}

__device__ __forceinline__ v8f wmma16(v16h a, v16h b, v8f c) {
  return __builtin_amdgcn_wmma_f32_16x16x32_f16(false, a, false, b,
                                                (short)0, c, false, false);
}

__device__ __forceinline__ float wave_sum(float v) {
#pragma unroll
  for (int off = 16; off > 0; off >>= 1) v += __shfl_xor(v, off, 32);
  return v;
}

// ---------------------------------------------------------------------------
// 1) Pack W^T (K x N view of an (N rows x K cols) f32 matrix) into
//    fragment-ready f16: dst[((nt*(K/32)+kb)*32 + lane)*16 + e]
//    with n = nt*16 + (lane&15), k = kb*32 + 16*(e>>3) + 8*(lane>>4) + (e&7).
// ---------------------------------------------------------------------------
__global__ void pack_b_kernel(const float* __restrict__ src,
                              h16_t* __restrict__ dst,
                              int N, int K, int srcStride, int colOff) {
  int idx = blockIdx.x * 256 + threadIdx.x;
  if (idx >= N * K) return;
  int e    = idx & 15;
  int lane = (idx >> 4) & 31;
  int kb   = (idx >> 9) % (K >> 5);
  int nt   = idx / ((K >> 5) << 9);
  int n = nt * 16 + (lane & 15);
  int k = kb * 32 + ((e >> 3) << 4) + ((lane >> 4) << 3) + (e & 7);
  dst[idx] = (h16_t)src[(size_t)n * srcStride + colOff + k];
}

// ---------------------------------------------------------------------------
// 2) Agent attention: one wave per (b,t).
// ---------------------------------------------------------------------------
__global__ __launch_bounds__(256)
void agent_attn_kernel(const float* __restrict__ x,
                       const float* __restrict__ Wa,
                       const float* __restrict__ ba,
                       const float* __restrict__ va,
                       float* __restrict__ xtRepr) {
  int tid = threadIdx.x, lane = tid & 31, wave = tid >> 5;
  size_t w = (size_t)blockIdx.x * 8 + wave;   // flat (b*T + t)
  const float* xp = x + w * 9;
  float xr[9];
#pragma unroll
  for (int f = 0; f < 9; ++f) xr[f] = xp[f];
  float sc0 = 0.f, sc1 = 0.f, sc2 = 0.f;
#pragma unroll
  for (int i = 0; i < 8; ++i) {
    int hh = lane + 32 * i;
    float wa0 = Wa[hh * 3], wa1 = Wa[hh * 3 + 1], wa2 = Wa[hh * 3 + 2];
    float bb = ba[hh], vv = va[hh];
    sc0 += vv * tanhf(wa0 * xr[0] + wa1 * xr[1] + wa2 * xr[2] + bb);
    sc1 += vv * tanhf(wa0 * xr[3] + wa1 * xr[4] + wa2 * xr[5] + bb);
    sc2 += vv * tanhf(wa0 * xr[6] + wa1 * xr[7] + wa2 * xr[8] + bb);
  }
  sc0 = wave_sum(sc0); sc1 = wave_sum(sc1); sc2 = wave_sum(sc2);
  float mx = fmaxf(sc0, fmaxf(sc1, sc2));
  float e0 = __expf(sc0 - mx), e1 = __expf(sc1 - mx), e2 = __expf(sc2 - mx);
  float inv = 1.f / (e0 + e1 + e2);
  if (lane < 3) {
    int f = lane;
    xtRepr[w * 3 + f] = inv * (e0 * xr[f] + e1 * xr[3 + f] + e2 * xr[6 + f]);
  }
}

// ---------------------------------------------------------------------------
// 3) Encoder GRU. 64 blocks x 256 threads; each block owns 16 batch rows.
// ---------------------------------------------------------------------------
__global__ __launch_bounds__(256)
void encoder_kernel(const float* __restrict__ xtRepr,
                    const float* __restrict__ Wih,  // (768,3)
                    const float* __restrict__ bih,
                    const float* __restrict__ bhh,
                    const h16_t* __restrict__ WhhP, // packed (K=256,N=768)
                    h16_t* __restrict__ encOut16,
                    float* __restrict__ hEnc) {
  extern __shared__ char smem[];
  h16_t* h16   = (h16_t*)smem;                          // 16 x 264
  float* h32   = (float*)(smem + BM * 264 * 2);         // 16 x 256
  float* gates = h32 + BM * HD;                         // 16 x 768
  float* xts   = gates + BM * G3;                       // 16 x 3

  int tid = threadIdx.x, lane = tid & 31, wave = tid >> 5;
  int grp = lane >> 4, r16 = lane & 15;
  int bbase = blockIdx.x * BM;

  for (int i = tid; i < BM * 264; i += 256) h16[i] = (h16_t)0.f;
  for (int i = tid; i < BM * HD;  i += 256) h32[i] = 0.f;

  // hoist per-column (j = tid) input weights / biases
  int j = tid;
  float w0[3], w1[3], w2[3];
#pragma unroll
  for (int f = 0; f < 3; ++f) {
    w0[f] = Wih[(size_t)j * 3 + f];
    w1[f] = Wih[(size_t)(HD + j) * 3 + f];
    w2[f] = Wih[(size_t)(2 * HD + j) * 3 + f];
  }
  float bi0 = bih[j], bi1 = bih[HD + j], bi2 = bih[2 * HD + j];
  float bh0 = bhh[j], bh1 = bhh[HD + j], bh2 = bhh[2 * HD + j];
  __syncthreads();

  for (int t = 0; t < TT; ++t) {
    // gates = h @ Whh^T  (16x256 @ 256x768); 8 waves x 6 N-tiles
    for (int nti = 0; nti < 6; ++nti) {
      int nt = wave * 6 + nti;
      v8f acc = {};
      const h16_t* bp = WhhP + (size_t)nt * 8 * 512;
#pragma unroll
      for (int kb = 0; kb < 8; ++kb) {
        v16h a = load_a(h16, 264, r16, kb, grp);
        v16h b = *(const v16h*)(bp + (kb * 32 + lane) * 16);
        acc = wmma16(a, b, acc);
      }
      int row0 = grp ? 8 : 0;
#pragma unroll
      for (int v = 0; v < 8; ++v)
        gates[(row0 + v) * G3 + nt * 16 + r16] = acc[v];
    }
    __syncthreads();
    if (tid < BM * 3) {
      int b = tid / 3, f = tid % 3;
      xts[b * 3 + f] = xtRepr[((size_t)(bbase + b) * TT + t) * 3 + f];
    }
    __syncthreads();
    // GRU pointwise: thread j handles column j for all 16 rows
    for (int b = 0; b < BM; ++b) {
      float x0 = xts[b * 3], x1 = xts[b * 3 + 1], x2 = xts[b * 3 + 2];
      float i0 = w0[0] * x0 + w0[1] * x1 + w0[2] * x2 + bi0;
      float i1 = w1[0] * x0 + w1[1] * x1 + w1[2] * x2 + bi1;
      float i2 = w2[0] * x0 + w2[1] * x1 + w2[2] * x2 + bi2;
      float g0 = gates[b * G3 + j] + bh0;
      float g1 = gates[b * G3 + HD + j] + bh1;
      float g2 = gates[b * G3 + 2 * HD + j] + bh2;
      float r = 1.f / (1.f + __expf(-(i0 + g0)));
      float z = 1.f / (1.f + __expf(-(i1 + g1)));
      float n = tanhf(i2 + r * g2);
      float hv = (1.f - z) * n + z * h32[b * HD + j];
      h32[b * HD + j] = hv;
      h16[b * 264 + j] = (h16_t)hv;
      encOut16[((size_t)(bbase + b) * TT + t) * HD + j] = (h16_t)hv;
    }
    __syncthreads();
  }
  for (int b = 0; b < BM; ++b)
    hEnc[(size_t)(bbase + b) * HD + tid] = h32[b * HD + tid];
}

// ---------------------------------------------------------------------------
// 4) enc_proj = enc_out @ Wt_enc^T  (M = B*T rows)
// ---------------------------------------------------------------------------
__global__ __launch_bounds__(256)
void encproj_kernel(const h16_t* __restrict__ encOut16,
                    const h16_t* __restrict__ WtEncP,
                    h16_t* __restrict__ encProj16) {
  int tid = threadIdx.x, lane = tid & 31, wave = tid >> 5;
  int grp = lane >> 4, r16 = lane & 15;
  size_t mbase = (size_t)blockIdx.x * 16;

  v16h afrag[8];
#pragma unroll
  for (int kb = 0; kb < 8; ++kb)
    afrag[kb] = load_a(encOut16 + mbase * HD, HD, r16, kb, grp);

  for (int nti = 0; nti < 2; ++nti) {
    int nt = wave * 2 + nti;
    v8f acc = {};
    const h16_t* bp = WtEncP + (size_t)nt * 8 * 512;
#pragma unroll
    for (int kb = 0; kb < 8; ++kb) {
      v16h b = *(const v16h*)(bp + (kb * 32 + lane) * 16);
      acc = wmma16(afrag[kb], b, acc);
    }
    int row0 = grp ? 8 : 0;
#pragma unroll
    for (int v = 0; v < 8; ++v)
      encProj16[(mbase + row0 + v) * HD + nt * 16 + r16] = (h16_t)acc[v];
  }
}

// ---------------------------------------------------------------------------
// 5) Decoder: 64 blocks x 256 threads; each block owns 16 batch rows for all
//    16 steps (fully independent across batch).
// ---------------------------------------------------------------------------
__global__ __launch_bounds__(256)
void decoder_kernel(const float* __restrict__ x,
                    const float* __restrict__ hEnc,
                    const h16_t* __restrict__ encOut16,
                    const h16_t* __restrict__ encProj16,
                    const h16_t* __restrict__ WtDecP,
                    const float* __restrict__ bt,
                    const float* __restrict__ vt,
                    const float* __restrict__ decWih,  // (768,265)
                    const float* __restrict__ decBih,
                    const float* __restrict__ decBhh,
                    const h16_t* __restrict__ decWihCtxP,
                    const h16_t* __restrict__ decWhhP,
                    const float* __restrict__ Wfc,
                    const float* __restrict__ bfc,
                    float* __restrict__ out) {
  extern __shared__ char smem[];
  h16_t* hs16  = (h16_t*)smem;                              // 16 x 264
  h16_t* ctx16 = hs16 + BM * 264;                           // 16 x 264
  float* h32   = (float*)(smem + 2 * BM * 264 * 2);         // 16 x 256
  float* dproj = h32 + BM * HD;                             // 16 x 256
  float* gI    = dproj + BM * HD;                           // 16 x 768
  float* gH    = gI + BM * G3;                              // 16 x 768
  float* din   = gH + BM * G3;                              // 16 x 12
  float* vts   = din + BM * 12;                             // 256
  float* bts   = vts + HD;                                  // 256

  int tid = threadIdx.x, lane = tid & 31, wave = tid >> 5;
  int grp = lane >> 4, r16 = lane & 15;
  int bbase = blockIdx.x * BM;

  vts[tid] = vt[tid];
  bts[tid] = bt[tid];
  for (int b = 0; b < BM; ++b) {
    float hv = hEnc[(size_t)(bbase + b) * HD + tid];
    h32[b * HD + tid] = hv;
    hs16[b * 264 + tid] = (h16_t)hv;
  }
  if (tid < BM * 9) {
    int b = tid / 9, o = tid % 9;
    din[b * 12 + o] = x[((size_t)(bbase + b) * TT + (TT - 1)) * 9 + o];
  }
  // hoist per-column decoder input weights / biases (j = tid)
  int j = tid;
  float wi0[9], wi1[9], wi2[9];
#pragma unroll
  for (int f = 0; f < 9; ++f) {
    wi0[f] = decWih[(size_t)j * 265 + f];
    wi1[f] = decWih[(size_t)(HD + j) * 265 + f];
    wi2[f] = decWih[(size_t)(2 * HD + j) * 265 + f];
  }
  float bi0 = decBih[j], bi1 = decBih[HD + j], bi2 = decBih[2 * HD + j];
  float bh0 = decBhh[j], bh1 = decBhh[HD + j], bh2 = decBhh[2 * HD + j];
  __syncthreads();

  for (int s = 0; s < PRED_LEN; ++s) {
    // --- (a) dproj = h @ Wt_dec^T + bt : 16 N-tiles, 2 per wave -----------
    for (int nti = 0; nti < 2; ++nti) {
      int nt = wave * 2 + nti;
      v8f acc = {};
      const h16_t* bp = WtDecP + (size_t)nt * 8 * 512;
#pragma unroll
      for (int kb = 0; kb < 8; ++kb) {
        v16h a = load_a(hs16, 264, r16, kb, grp);
        v16h b = *(const v16h*)(bp + (kb * 32 + lane) * 16);
        acc = wmma16(a, b, acc);
      }
      int row0 = grp ? 8 : 0;
#pragma unroll
      for (int v = 0; v < 8; ++v)
        dproj[(row0 + v) * HD + nt * 16 + r16] = acc[v] + bts[nt * 16 + r16];
    }
    __syncthreads();
    // --- (b) temporal attention, online softmax; wave -> 2 rows ----------
    for (int rr = 0; rr < 2; ++rr) {
      int b = wave * 2 + rr;
      const h16_t* epRow = encProj16 + (size_t)(bbase + b) * TT * HD;
      const h16_t* eoRow = encOut16 + (size_t)(bbase + b) * TT * HD;
      int h0 = lane * 8;
      float dp[8], vv[8], cacc[8];
#pragma unroll
      for (int i = 0; i < 8; ++i) {
        dp[i] = dproj[b * HD + h0 + i];
        vv[i] = vts[h0 + i];
        cacc[i] = 0.f;
      }
      float m = -1e30f, l = 0.f;
      for (int t = 0; t < TT; ++t) {
        v8h ep = *(const v8h*)(epRow + (size_t)t * HD + h0);
        float sp = 0.f;
#pragma unroll
        for (int i = 0; i < 8; ++i)
          sp += vv[i] * tanhf((float)ep[i] + dp[i]);
        sp = wave_sum(sp);
        float mn = fmaxf(m, sp);
        float sc = __expf(m - mn);
        float wg = __expf(sp - mn);
        l = l * sc + wg;
        v8h eo = *(const v8h*)(eoRow + (size_t)t * HD + h0);
#pragma unroll
        for (int i = 0; i < 8; ++i)
          cacc[i] = cacc[i] * sc + wg * (float)eo[i];
        m = mn;
      }
      float inv = 1.f / l;
      v8h cv;
#pragma unroll
      for (int i = 0; i < 8; ++i) cv[i] = (h16_t)(cacc[i] * inv);
      *(v8h*)(ctx16 + b * 264 + h0) = cv;
    }
    __syncthreads();
    // --- (c) gI = ctx @ Wih_ctx^T ; gH = h @ Whh^T : 48 N-tiles each ------
    for (int nti = 0; nti < 6; ++nti) {
      int nt = wave * 6 + nti;
      int row0 = grp ? 8 : 0;
      v8f acc = {};
      const h16_t* bp = decWihCtxP + (size_t)nt * 8 * 512;
#pragma unroll
      for (int kb = 0; kb < 8; ++kb) {
        v16h a = load_a(ctx16, 264, r16, kb, grp);
        v16h b = *(const v16h*)(bp + (kb * 32 + lane) * 16);
        acc = wmma16(a, b, acc);
      }
#pragma unroll
      for (int v = 0; v < 8; ++v)
        gI[(row0 + v) * G3 + nt * 16 + r16] = acc[v];
      acc = (v8f){};
      bp = decWhhP + (size_t)nt * 8 * 512;
#pragma unroll
      for (int kb = 0; kb < 8; ++kb) {
        v16h a = load_a(hs16, 264, r16, kb, grp);
        v16h b = *(const v16h*)(bp + (kb * 32 + lane) * 16);
        acc = wmma16(a, b, acc);
      }
#pragma unroll
      for (int v = 0; v < 8; ++v)
        gH[(row0 + v) * G3 + nt * 16 + r16] = acc[v];
    }
    __syncthreads();
    // --- (d) GRU pointwise: thread j, 16 rows -----------------------------
    for (int b = 0; b < BM; ++b) {
      float i0 = bi0 + gI[b * G3 + j];
      float i1 = bi1 + gI[b * G3 + HD + j];
      float i2 = bi2 + gI[b * G3 + 2 * HD + j];
#pragma unroll
      for (int f = 0; f < 9; ++f) {
        float xv = din[b * 12 + f];
        i0 += wi0[f] * xv; i1 += wi1[f] * xv; i2 += wi2[f] * xv;
      }
      float g0 = gH[b * G3 + j] + bh0;
      float g1 = gH[b * G3 + HD + j] + bh1;
      float g2 = gH[b * G3 + 2 * HD + j] + bh2;
      float r = 1.f / (1.f + __expf(-(i0 + g0)));
      float z = 1.f / (1.f + __expf(-(i1 + g1)));
      float n = tanhf(i2 + r * g2);
      float hv = (1.f - z) * n + z * h32[b * HD + j];
      h32[b * HD + j] = hv;
      hs16[b * 264 + j] = (h16_t)hv;
    }
    __syncthreads();
    // --- (e) pred = h_new @ Wfc^T + bfc ; feed back as next dec_in --------
    if (tid < BM * 9) {
      int b = tid / 9, o = tid % 9;
      float acc = bfc[o];
      for (int hh = 0; hh < HD; ++hh)
        acc += Wfc[o * HD + hh] * h32[b * HD + hh];
      out[((size_t)(bbase + b) * PRED_LEN + s) * 9 + o] = acc;
      din[b * 12 + o] = acc;
    }
    __syncthreads();
  }
}

// ---------------------------------------------------------------------------
extern "C" void kernel_launch(void* const* d_in, const int* in_sizes, int n_in,
                              void* d_out, int out_size, void* d_ws,
                              size_t ws_size, hipStream_t stream) {
  const float* x       = (const float*)d_in[0];
  const float* Wa      = (const float*)d_in[1];
  const float* ba      = (const float*)d_in[2];
  const float* va      = (const float*)d_in[3];
  const float* encWih  = (const float*)d_in[4];
  const float* encWhh  = (const float*)d_in[5];
  const float* encBih  = (const float*)d_in[6];
  const float* encBhh  = (const float*)d_in[7];
  const float* decWih  = (const float*)d_in[8];
  const float* decWhh  = (const float*)d_in[9];
  const float* decBih  = (const float*)d_in[10];
  const float* decBhh  = (const float*)d_in[11];
  const float* Wt      = (const float*)d_in[12];
  const float* btv     = (const float*)d_in[13];
  const float* vtv     = (const float*)d_in[14];
  const float* Wfc     = (const float*)d_in[15];
  const float* bfc     = (const float*)d_in[16];
  float* out           = (float*)d_out;

  char* ws = (char*)d_ws;
  const size_t B = 1024, T = 256, H = 256;
  size_t off = 0;
  float* xtRepr    = (float*)(ws + off); off += B * T * 3 * 4;        // 3 MB
  h16_t* encOut16  = (h16_t*)(ws + off); off += B * T * H * 2;        // 128 MB
  h16_t* encProj16 = (h16_t*)(ws + off); off += B * T * H * 2;        // 128 MB
  float* hEnc      = (float*)(ws + off); off += B * H * 4;            // 1 MB
  h16_t* pWhh      = (h16_t*)(ws + off); off += 768 * 256 * 2;
  h16_t* pWtEnc    = (h16_t*)(ws + off); off += 256 * 256 * 2;
  h16_t* pWtDec    = (h16_t*)(ws + off); off += 256 * 256 * 2;
  h16_t* pDecWhh   = (h16_t*)(ws + off); off += 768 * 256 * 2;
  h16_t* pDecWihC  = (h16_t*)(ws + off); off += 768 * 256 * 2;

  // 1) pack weights (B-matrix fragment layout, f16)
  pack_b_kernel<<<768, 256, 0, stream>>>(encWhh, pWhh, 768, 256, 256, 0);
  pack_b_kernel<<<256, 256, 0, stream>>>(Wt, pWtEnc, 256, 256, 512, 256);
  pack_b_kernel<<<256, 256, 0, stream>>>(Wt, pWtDec, 256, 256, 512, 0);
  pack_b_kernel<<<768, 256, 0, stream>>>(decWhh, pDecWhh, 768, 256, 256, 0);
  pack_b_kernel<<<768, 256, 0, stream>>>(decWih, pDecWihC, 768, 256, 265, 9);

  // 2) agent attention
  agent_attn_kernel<<<(B * T) / 8, 256, 0, stream>>>(x, Wa, ba, va, xtRepr);

  // 3) encoder GRU (64 blocks own 16 batch rows each, full T loop)
  size_t encLds = BM * 264 * 2 + BM * HD * 4 + BM * G3 * 4 + BM * 3 * 4;
  encoder_kernel<<<B / BM, 256, encLds, stream>>>(xtRepr, encWih, encBih,
                                                  encBhh, pWhh, encOut16,
                                                  hEnc);

  // 4) enc_proj GEMM
  encproj_kernel<<<(B * T) / 16, 256, 0, stream>>>(encOut16, pWtEnc,
                                                   encProj16);

  // 5) decoder (16 steps per block)
  size_t decLds = 2 * BM * 264 * 2 + 2 * BM * HD * 4 + 2 * BM * G3 * 4 +
                  BM * 12 * 4 + 2 * HD * 4;
  decoder_kernel<<<B / BM, 256, decLds, stream>>>(
      x, hEnc, encOut16, encProj16, pWtDec, btv, vtv, decWih, decBih, decBhh,
      pDecWihC, pDecWhh, Wfc, bfc, out);
}
